// NTSNet_6983616823584
// MI455X (gfx1250) — compile-verified, matrix operand order checked by reading, not run on hardware
//
#include <hip/hip_runtime.h>

// ---------------------------------------------------------------------------
// NTS-Net forward for MI455X (gfx1250, wave32).
//  * 2x patchify-conv GEMMs (3136 x 3072 x 2048)  -> v_wmma_f32_16x16x32_bf16
//  * navigator conv1 (3136 x 9*2048 x 128)        -> v_wmma_f32_16x16x32_bf16
// fp32 accumulate, bf16 operands staged through LDS (v_cvt_pk_bf16_f32).
// Backbone activations kept NHWC so channels are contiguous for implicit GEMM
// and all WMMA epilogue stores are lane-coalesced.
// ---------------------------------------------------------------------------

#define B_SZ        16
#define TOPN        4
#define NUM_ANCH    1614
#define NUM_CLASSES 200
#define NCH         2048          // backbone output channels
#define KDIM        3072          // 3*32*32 patch K
#define OUTR        224
#define IMG         448
#define PADP        224
#define IOU_TH      0.25f

typedef __attribute__((ext_vector_type(16))) __bf16 v16bf;
typedef __attribute__((ext_vector_type(8)))  __bf16 v8bf;
typedef __attribute__((ext_vector_type(4)))  __bf16 v4bf;
typedef __attribute__((ext_vector_type(8)))  float  v8f;

// ---------------------------------------------------------------------------
// Patchify-conv as WMMA GEMM (stride-32/kernel-32 conv == dense GEMM).
//   A[p][k] = X[b][c][oy*32+ky][ox*32+kx]   (p = b*PPI + oy*OW + ox)
//   B[k][n] = W[n][k]                       (bb_w is [N][K], K-contiguous)
//   Out[p][n] = sum_k A*B + bias[n]         (NHWC: n contiguous)
// Block: 256 threads = 8 waves; tile 64(M) x 64(N); K stepped by 32.
// K-slices of 32 are contiguous 128B global segments for both A and B.
// ---------------------------------------------------------------------------
__global__ __launch_bounds__(256)
void patch_gemm_wmma(const float* __restrict__ X, const float* __restrict__ W,
                     const float* __restrict__ bias, float* __restrict__ Out,
                     int H, int OW, int PPI)
{
    __shared__ __align__(16) __bf16 lA[64][40];   // 40 = 32 + pad (bank spread)
    __shared__ __align__(16) __bf16 lB[64][40];

    const int tid = threadIdx.x;
    const int mt  = blockIdx.x;   // M tile (64 output positions)
    const int nt  = blockIdx.y;   // N tile (64 output channels)

    // ---- loader roles: 256 threads cover 64 rows x 32 floats (8 each)
    const int lr = tid >> 2;      // row in tile 0..63
    const int lq = tid & 3;       // float group [lq*8, lq*8+8)

    const int p   = mt * 64 + lr; // global M row (grid covers M exactly)
    const int pb  = p / PPI;
    const int ps  = p - pb * PPI;
    const int oy  = ps / OW;
    const int ox  = ps - oy * OW;
    const int nrw = nt * 64 + lr; // global N row for B loads

    // ---- compute roles: wave w -> A sub-tile (w&3), B sub-tiles (w>>2)*2, +1
    const int wv   = tid >> 5;
    const int lane = tid & 31;
    const int mi   = wv & 3;
    const int nb   = (wv >> 2) * 2;
    const int fr   = lane & 15;
    const int kb   = (lane < 16) ? 0 : 8;   // 16-bit A/B frag: K base per half

    v8f acc0 = {};
    v8f acc1 = {};

    for (int kk = 0; kk < KDIM / 32; ++kk) {
        const int k0 = kk * 32;
        const int c  = k0 >> 10;          // k = c*1024 + ky*32 + kx
        const int ky = (k0 >> 5) & 31;

        const float* ap = X + (((size_t)(pb * 3 + c) * H + (oy * 32 + ky)) * H
                               + ox * 32 + lq * 8);
        const float* bp = W + ((size_t)nrw * KDIM + k0 + lq * 8);

        if (kk + 1 < KDIM / 32) {         // global_prefetch_b8 next K slice
            const int k1  = k0 + 32;
            const int c1  = k1 >> 10;
            const int ky1 = (k1 >> 5) & 31;
            __builtin_prefetch(X + (((size_t)(pb * 3 + c1) * H + (oy * 32 + ky1)) * H
                                    + ox * 32 + lq * 8), 0, 1);
            __builtin_prefetch(bp + 32, 0, 1);
        }

        const float4 a0 = reinterpret_cast<const float4*>(ap)[0];
        const float4 a1 = reinterpret_cast<const float4*>(ap)[1];
        const float4 b0 = reinterpret_cast<const float4*>(bp)[0];
        const float4 b1 = reinterpret_cast<const float4*>(bp)[1];

        __syncthreads();                  // previous iter done reading LDS

        v8bf pa, pbv;
        pa[0]=(__bf16)a0.x; pa[1]=(__bf16)a0.y; pa[2]=(__bf16)a0.z; pa[3]=(__bf16)a0.w;
        pa[4]=(__bf16)a1.x; pa[5]=(__bf16)a1.y; pa[6]=(__bf16)a1.z; pa[7]=(__bf16)a1.w;
        pbv[0]=(__bf16)b0.x; pbv[1]=(__bf16)b0.y; pbv[2]=(__bf16)b0.z; pbv[3]=(__bf16)b0.w;
        pbv[4]=(__bf16)b1.x; pbv[5]=(__bf16)b1.y; pbv[6]=(__bf16)b1.z; pbv[7]=(__bf16)b1.w;
        *reinterpret_cast<v8bf*>(&lA[lr][lq * 8]) = pa;    // 16B ds write
        *reinterpret_cast<v8bf*>(&lB[lr][lq * 8]) = pbv;

        __syncthreads();

        // 16-bit A/B frag layout: lane<16 K={0..7,16..23}; lane>=16 K={8..15,24..31}
        v16bf af, bf0, bf1;
        #pragma unroll
        for (int i = 0; i < 8; ++i) {
            af [i]     = lA[mi * 16 + fr][kb + i];
            af [i + 8] = lA[mi * 16 + fr][16 + kb + i];
            bf0[i]     = lB[nb * 16 + fr][kb + i];
            bf0[i + 8] = lB[nb * 16 + fr][16 + kb + i];
            bf1[i]     = lB[(nb + 1) * 16 + fr][kb + i];
            bf1[i + 8] = lB[(nb + 1) * 16 + fr][16 + kb + i];
        }
        acc0 = __builtin_amdgcn_wmma_f32_16x16x32_bf16(false, af, false, bf0,
                                                       (short)0, acc0, false, false);
        acc1 = __builtin_amdgcn_wmma_f32_16x16x32_bf16(false, af, false, bf1,
                                                       (short)0, acc1, false, false);
    }

    // Epilogue (NHWC): lane holds column n = fr -> coalesced 4B stores
    #pragma unroll
    for (int t = 0; t < 2; ++t) {
        const int gn  = nt * 64 + (nb + t) * 16 + fr;
        const float bv = bias[gn];
        const v8f acc = t ? acc1 : acc0;
        #pragma unroll
        for (int vi = 0; vi < 8; ++vi) {
            const int ml = vi + ((lane < 16) ? 0 : 8);
            const int gm = mt * 64 + mi * 16 + ml;
            Out[(size_t)gm * NCH + gn] = acc[vi] + bv;
        }
    }
}

// ---------------------------------------------------------------------------
// n1_dw [128][2048][3][3] -> wT[t][n][c] (c contiguous) for tap-GEMM B loads.
// ---------------------------------------------------------------------------
__global__ void transpose_n1w_kernel(const float* __restrict__ w, float* __restrict__ wT)
{
    int gid = blockIdx.x * blockDim.x + threadIdx.x;
    const int total = 128 * 2048 * 9;
    if (gid >= total) return;
    int t = gid % 9; int r = gid / 9;
    int c = r % 2048; int n = r / 2048;
    wT[((size_t)t * 128 + n) * 2048 + c] = w[gid];
}

// ---------------------------------------------------------------------------
// Navigator conv1 (2048->128, 3x3, pad 1, 14x14) as 9 tap-GEMMs on WMMA.
//   M = 3136 positions (NHWC raw), N = 128, K = 9 * 2048.
// Block: 256 threads = 8 waves; tile 32(M) x 128(N); K stepped by 32.
// Border taps zero-filled per A row. Output y1 in NCHW (+bias, ReLU).
// ---------------------------------------------------------------------------
__global__ __launch_bounds__(256)
void nav_conv1_wmma(const float* __restrict__ rawT, const float* __restrict__ wT,
                    const float* __restrict__ bias, float* __restrict__ y1)
{
    __shared__ __align__(16) __bf16 lA[32][40];
    __shared__ __align__(16) __bf16 lB[128][40];

    const int tid = threadIdx.x;
    const int mt  = blockIdx.x;              // 98 tiles of 32 rows

    // A loader: 256 threads cover 32 rows x 32 floats (4 each)
    const int ra = tid >> 3;                 // 0..31
    const int qa = tid & 7;                  // float group [qa*4, qa*4+4)
    const int pA = mt * 32 + ra;
    const int ab = pA / 196;
    const int as = pA - ab * 196;
    const int ay = as / 14;
    const int ax = as - ay * 14;

    // B loader: 256 threads cover 128 rows x 32 floats (16 each)
    const int rb = tid >> 1;                 // 0..127
    const int qb = tid & 1;                  // float group [qb*16, qb*16+16)

    // compute roles: wave w -> A sub-tile (w&1), B sub-tiles ((w>>1)*2, +1)
    const int wv   = tid >> 5;
    const int lane = tid & 31;
    const int mi   = wv & 1;
    const int nb   = (wv >> 1) * 2;
    const int fr   = lane & 15;
    const int kb   = (lane < 16) ? 0 : 8;

    v8f acc0 = {};
    v8f acc1 = {};

    for (int t = 0; t < 9; ++t) {
        const int dy = t / 3 - 1, dx = t % 3 - 1;
        const int iy = ay + dy,   ix = ax + dx;
        const bool valid = (unsigned)iy < 14u && (unsigned)ix < 14u;
        const float* arow = rawT + ((size_t)(ab * 196 + iy * 14 + ix)) * NCH;
        const float* brow = wT + ((size_t)t * 128 + rb) * NCH;

        for (int kc = 0; kc < NCH / 32; ++kc) {
            const int c0 = kc * 32;

            float4 av = {0.f, 0.f, 0.f, 0.f};
            if (valid) av = reinterpret_cast<const float4*>(arow + c0 + qa * 4)[0];
            const float4 b0 = reinterpret_cast<const float4*>(brow + c0 + qb * 16)[0];
            const float4 b1 = reinterpret_cast<const float4*>(brow + c0 + qb * 16)[1];
            const float4 b2 = reinterpret_cast<const float4*>(brow + c0 + qb * 16)[2];
            const float4 b3 = reinterpret_cast<const float4*>(brow + c0 + qb * 16)[3];
            if (kc + 1 < NCH / 32)
                __builtin_prefetch(brow + c0 + 32 + qb * 16, 0, 1);

            __syncthreads();              // previous iter done reading LDS

            v4bf pa;
            pa[0]=(__bf16)av.x; pa[1]=(__bf16)av.y; pa[2]=(__bf16)av.z; pa[3]=(__bf16)av.w;
            v8bf p0, p1;
            p0[0]=(__bf16)b0.x; p0[1]=(__bf16)b0.y; p0[2]=(__bf16)b0.z; p0[3]=(__bf16)b0.w;
            p0[4]=(__bf16)b1.x; p0[5]=(__bf16)b1.y; p0[6]=(__bf16)b1.z; p0[7]=(__bf16)b1.w;
            p1[0]=(__bf16)b2.x; p1[1]=(__bf16)b2.y; p1[2]=(__bf16)b2.z; p1[3]=(__bf16)b2.w;
            p1[4]=(__bf16)b3.x; p1[5]=(__bf16)b3.y; p1[6]=(__bf16)b3.z; p1[7]=(__bf16)b3.w;
            *reinterpret_cast<v4bf*>(&lA[ra][qa * 4])      = pa;   // 8B ds write
            *reinterpret_cast<v8bf*>(&lB[rb][qb * 16])     = p0;   // 16B ds writes
            *reinterpret_cast<v8bf*>(&lB[rb][qb * 16 + 8]) = p1;

            __syncthreads();

            v16bf af, bf0, bf1;
            #pragma unroll
            for (int i = 0; i < 8; ++i) {
                af [i]     = lA[mi * 16 + fr][kb + i];
                af [i + 8] = lA[mi * 16 + fr][16 + kb + i];
                bf0[i]     = lB[nb * 16 + fr][kb + i];
                bf0[i + 8] = lB[nb * 16 + fr][16 + kb + i];
                bf1[i]     = lB[(nb + 1) * 16 + fr][kb + i];
                bf1[i + 8] = lB[(nb + 1) * 16 + fr][16 + kb + i];
            }
            acc0 = __builtin_amdgcn_wmma_f32_16x16x32_bf16(false, af, false, bf0,
                                                           (short)0, acc0, false, false);
            acc1 = __builtin_amdgcn_wmma_f32_16x16x32_bf16(false, af, false, bf1,
                                                           (short)0, acc1, false, false);
        }
    }

    // Epilogue: y1 NCHW = relu(acc + bias)
    #pragma unroll
    for (int t = 0; t < 2; ++t) {
        const int gn = (nb + t) * 16 + fr;           // 0..127
        const float bv = bias[gn];
        const v8f acc = t ? acc1 : acc0;
        #pragma unroll
        for (int vi = 0; vi < 8; ++vi) {
            const int ml = vi + ((lane < 16) ? 0 : 8);
            const int gm = mt * 32 + mi * 16 + ml;
            const int gb = gm / 196;
            const int gs = gm - gb * 196;
            y1[((size_t)gb * 128 + gn) * 196 + gs] = fmaxf(acc[vi] + bv, 0.f);
        }
    }
}

// ---------------------------------------------------------------------------
// Small navigator 3x3 convs (128ch) + bias + ReLU, thread-per-output (NCHW).
// ---------------------------------------------------------------------------
__global__ void conv3x3_relu_kernel(const float* __restrict__ in, const float* __restrict__ w,
                                    const float* __restrict__ bias, float* __restrict__ out,
                                    int Cin, int Hin, int Cout, int Hout, int stride, int total)
{
    int gid = blockIdx.x * blockDim.x + threadIdx.x;
    if (gid >= total) return;
    int ox = gid % Hout; int t = gid / Hout;
    int oy = t % Hout;   t /= Hout;
    int co = t % Cout;   int b = t / Cout;

    float acc = bias[co];
    for (int c = 0; c < Cin; ++c) {
        const float* ip = in + ((size_t)(b * Cin + c) * Hin) * Hin;
        const float* wp = w + (size_t)(co * Cin + c) * 9;
        #pragma unroll
        for (int ky = 0; ky < 3; ++ky) {
            int iy = oy * stride + ky - 1;
            if (iy < 0 || iy >= Hin) continue;
            #pragma unroll
            for (int kx = 0; kx < 3; ++kx) {
                int ix = ox * stride + kx - 1;
                if (ix < 0 || ix >= Hin) continue;
                acc += ip[iy * Hin + ix] * wp[ky * 3 + kx];
            }
        }
    }
    out[gid] = fmaxf(acc, 0.f);
}

// ---------------------------------------------------------------------------
// 1x1 score heads; writes concat layout: out[b*outStride + co*HW + s].
// ---------------------------------------------------------------------------
__global__ void conv1x1_kernel(const float* __restrict__ in, const float* __restrict__ w,
                               const float* __restrict__ bias, float* __restrict__ out,
                               int Cin, int HW, int Cout, int outStride, int total)
{
    int gid = blockIdx.x * blockDim.x + threadIdx.x;
    if (gid >= total) return;
    int s  = gid % HW;  int t = gid / HW;
    int co = t % Cout;  int b = t / Cout;
    float acc = bias[co];
    const float* ip = in + (size_t)b * Cin * HW + s;
    const float* wp = w + (size_t)co * Cin;
    for (int c = 0; c < Cin; ++c) acc += ip[(size_t)c * HW] * wp[c];
    out[(size_t)b * outStride + co * HW + s] = acc;
}

// ---------------------------------------------------------------------------
// Greedy NMS, one workgroup per image. Mimics jnp.argmax first-max ties.
// ---------------------------------------------------------------------------
#define NMS_T 256
#define NMS_PER 7   // ceil(1614/256)

__global__ void nms_kernel(const float* __restrict__ rpn, const int* __restrict__ anchors,
                           int* __restrict__ top_idx)
{
    const int b   = blockIdx.x;
    const int tid = threadIdx.x;
    const float* sc = rpn + (size_t)b * NUM_ANCH;

    float sv[NMS_PER];
    bool  vv[NMS_PER];
    #pragma unroll
    for (int i = 0; i < NMS_PER; ++i) {
        int a = tid + i * NMS_T;
        if (a < NUM_ANCH) { sv[i] = sc[a]; vv[i] = true; }
        else              { sv[i] = -__builtin_inff(); vv[i] = false; }
    }

    __shared__ float rs[NMS_T];
    __shared__ int   ri[NMS_T];
    __shared__ float pb[4];
    __shared__ float parea;

    for (int pick = 0; pick < TOPN; ++pick) {
        float best = -__builtin_inff();
        int   bi   = 0x7FFFFFFF;
        #pragma unroll
        for (int i = 0; i < NMS_PER; ++i) {
            int a = tid + i * NMS_T;
            float s = vv[i] ? sv[i] : -__builtin_inff();
            if (s > best || (s == best && a < bi)) { best = s; bi = a; }
        }
        rs[tid] = best; ri[tid] = bi;
        __syncthreads();
        for (int off = NMS_T / 2; off > 0; off >>= 1) {
            if (tid < off) {
                if (rs[tid + off] > rs[tid] ||
                    (rs[tid + off] == rs[tid] && ri[tid + off] < ri[tid])) {
                    rs[tid] = rs[tid + off]; ri[tid] = ri[tid + off];
                }
            }
            __syncthreads();
        }
        if (tid == 0) {
            int w = ri[0];
            top_idx[b * TOPN + pick] = w;
            float y0 = (float)anchors[w*4+0], x0 = (float)anchors[w*4+1];
            float y1 = (float)anchors[w*4+2], x1 = (float)anchors[w*4+3];
            pb[0] = y0; pb[1] = x0; pb[2] = y1; pb[3] = x1;
            parea = (y1 - y0) * (x1 - x0);
        }
        __syncthreads();
        #pragma unroll
        for (int i = 0; i < NMS_PER; ++i) {
            int a = tid + i * NMS_T;
            if (a < NUM_ANCH && vv[i]) {
                float ay0=(float)anchors[a*4+0], ax0=(float)anchors[a*4+1];
                float ay1=(float)anchors[a*4+2], ax1=(float)anchors[a*4+3];
                float ih = fminf(ay1, pb[2]) - fmaxf(ay0, pb[0]);
                float iw = fminf(ax1, pb[3]) - fmaxf(ax0, pb[1]);
                float inter = (ih < 0.f || iw < 0.f) ? 0.f : ih * iw;
                float area  = (ay1 - ay0) * (ax1 - ax0);
                float iou   = inter / (area + parea - inter);
                if (!(iou < IOU_TH)) vv[i] = false;
            }
        }
        __syncthreads();
    }
}

// ---------------------------------------------------------------------------
// Bilinear crop-resize of selected (padded-coordinate) boxes -> (64,3,224,224).
// Padding emulated by zero-return outside the original 448x448 image.
// ---------------------------------------------------------------------------
__global__ void crop_resize_kernel(const float* __restrict__ x, const int* __restrict__ anchors,
                                   const int* __restrict__ top_idx, float* __restrict__ out)
{
    int gid = blockIdx.x * blockDim.x + threadIdx.x;
    const int total = B_SZ * TOPN * 3 * OUTR * OUTR;
    if (gid >= total) return;
    int ox = gid % OUTR; int t = gid / OUTR;
    int oy = t % OUTR;   t /= OUTR;
    int c  = t % 3;      int bt = t / 3;
    int b  = bt / TOPN;

    const int* box = anchors + (size_t)top_idx[bt] * 4;
    int y0 = box[0], x0 = box[1], y1 = box[2], x1 = box[3];
    float h = (float)(y1 - y0), w = (float)(x1 - x0);
    float sy = (float)oy * (h - 1.f) / (float)(OUTR - 1);
    float sx = (float)ox * (w - 1.f) / (float)(OUTR - 1);
    float fy = floorf(sy), fx = floorf(sx);
    float wy = sy - fy,    wx = sx - fx;
    int iy0 = y0 + (int)fy, ix0 = x0 + (int)fx;
    int iy1 = min(iy0 + 1, y1 - 1);
    int ix1 = min(ix0 + 1, x1 - 1);

    const float* img = x + ((size_t)b * 3 + c) * IMG * IMG;
    auto samp = [&](int py, int px) -> float {
        py -= PADP; px -= PADP;
        if (py < 0 || py >= IMG || px < 0 || px >= IMG) return 0.f;
        return img[py * IMG + px];
    };
    float v00 = samp(iy0, ix0), v01 = samp(iy0, ix1);
    float v10 = samp(iy1, ix0), v11 = samp(iy1, ix1);
    float top = v00 * (1.f - wx) + v01 * wx;
    float bot = v10 * (1.f - wx) + v11 * wx;
    out[gid] = top * (1.f - wy) + bot * wy;
}

// ---------------------------------------------------------------------------
// NHWC spatial mean: out[b*C+n] = mean_s in[(b*L+s)*C + n]  (coalesced in n).
// ---------------------------------------------------------------------------
__global__ void mean_nhwc_kernel(const float* __restrict__ in, float* __restrict__ out,
                                 int groups, int L, int C)
{
    int g = blockIdx.x * blockDim.x + threadIdx.x;
    if (g >= groups * C) return;
    int b = g / C;
    int n = g - b * C;
    const float* p = in + (size_t)b * L * C + n;
    float s = 0.f;
    for (int i = 0; i < L; ++i) s += p[(size_t)i * C];
    out[g] = s / (float)L;
}

// ---------------------------------------------------------------------------
// Classifier: logits[b][j] = concat[b] . cn_w[j] + cn_b[j]; one wave32/output.
// concat = [part_feats (t*2048+n), raw_feats] built on the fly.
// ---------------------------------------------------------------------------
__global__ void classifier_kernel(const float* __restrict__ pfeat, const float* __restrict__ rfeat,
                                  const float* __restrict__ cn_w, const float* __restrict__ cn_b,
                                  float* __restrict__ out)
{
    int gid  = blockIdx.x * blockDim.x + threadIdx.x;
    int wid  = gid >> 5;
    int lane = gid & 31;
    if (wid >= B_SZ * NUM_CLASSES) return;
    int b = wid / NUM_CLASSES;
    int j = wid - b * NUM_CLASSES;
    const float* wr = cn_w + (size_t)j * (TOPN * NCH + NCH);
    float sum = 0.f;
    for (int k = lane; k < TOPN * NCH + NCH; k += 32) {
        float a = (k < TOPN * NCH)
                    ? pfeat[((size_t)b * TOPN + (k >> 11)) * NCH + (k & (NCH - 1))]
                    : rfeat[(size_t)b * NCH + (k - TOPN * NCH)];
        sum += a * wr[k];
    }
    #pragma unroll
    for (int off = 16; off > 0; off >>= 1) sum += __shfl_down(sum, off, 32);
    if (lane == 0) out[(size_t)b * NUM_CLASSES + j] = sum + cn_b[j];
}

// ---------------------------------------------------------------------------
extern "C" void kernel_launch(void* const* d_in, const int* in_sizes, int n_in,
                              void* d_out, int out_size, void* d_ws, size_t ws_size,
                              hipStream_t stream)
{
    (void)in_sizes; (void)n_in; (void)out_size; (void)ws_size;
    const float* x     = (const float*)d_in[0];
    const float* bb_w  = (const float*)d_in[1];
    const float* bb_b  = (const float*)d_in[2];
    const float* n1_dw = (const float*)d_in[3];
    const float* n1_db = (const float*)d_in[4];
    const float* n1_tw = (const float*)d_in[5];
    const float* n1_tb = (const float*)d_in[6];
    const float* n2_dw = (const float*)d_in[7];
    const float* n2_db = (const float*)d_in[8];
    const float* n2_tw = (const float*)d_in[9];
    const float* n2_tb = (const float*)d_in[10];
    const float* n3_dw = (const float*)d_in[11];
    const float* n3_db = (const float*)d_in[12];
    const float* n3_tw = (const float*)d_in[13];
    const float* n3_tb = (const float*)d_in[14];
    const float* cn_w  = (const float*)d_in[15];
    const float* cn_b  = (const float*)d_in[16];
    const int*   anch  = (const int*)d_in[17];
    float* out = (float*)d_out;

    // ---- workspace carve-out (256B aligned); raw (NHWC) reused as part_raw
    char* ws = (char*)d_ws;
    auto alloc = [&](size_t bytes) -> char* {
        char* p = ws; ws += (bytes + 255) & ~((size_t)255); return p;
    };
    float* raw   = (float*)alloc((size_t)3136 * NCH * 4);       // NHWC [p][n]
    float* wT    = (float*)alloc((size_t)9 * 128 * NCH * 4);    // [tap][n][c]
    float* y1    = (float*)alloc((size_t)B_SZ * 128 * 196 * 4); // NCHW
    float* y2    = (float*)alloc((size_t)B_SZ * 128 * 49 * 4);
    float* y3    = (float*)alloc((size_t)B_SZ * 128 * 16 * 4);
    float* rpn   = (float*)alloc((size_t)B_SZ * NUM_ANCH * 4);
    int*   tidx  = (int*)  alloc((size_t)B_SZ * TOPN * 4);
    float* parts = (float*)alloc((size_t)B_SZ * TOPN * 3 * OUTR * OUTR * 4);
    float* pfeat = (float*)alloc((size_t)B_SZ * TOPN * NCH * 4);
    float* rfeat = (float*)alloc((size_t)B_SZ * NCH * 4);

    const dim3 gemm_grid(49, 32);   // M=3136/64, N=2048/64 (exact)

    // 1) backbone patchify conv on full images (NHWC out)
    patch_gemm_wmma<<<gemm_grid, 256, 0, stream>>>(x, bb_w, bb_b, raw, 448, 14, 196);

    // 2) navigator conv1 on WMMA (weight transpose first), then small heads
    { int tot = 128 * 2048 * 9;
      transpose_n1w_kernel<<<(tot + 255) / 256, 256, 0, stream>>>(n1_dw, wT); }
    nav_conv1_wmma<<<98, 256, 0, stream>>>(raw, wT, n1_db, y1);

    { int tot = B_SZ * 6 * 196;
      conv1x1_kernel<<<(tot + 255) / 256, 256, 0, stream>>>(y1, n1_tw, n1_tb, rpn + 0, 128, 196, 6, NUM_ANCH, tot); }
    { int tot = B_SZ * 128 * 7 * 7;
      conv3x3_relu_kernel<<<(tot + 255) / 256, 256, 0, stream>>>(y1, n2_dw, n2_db, y2, 128, 14, 128, 7, 2, tot); }
    { int tot = B_SZ * 6 * 49;
      conv1x1_kernel<<<(tot + 255) / 256, 256, 0, stream>>>(y2, n2_tw, n2_tb, rpn + 1176, 128, 49, 6, NUM_ANCH, tot); }
    { int tot = B_SZ * 128 * 4 * 4;
      conv3x3_relu_kernel<<<(tot + 255) / 256, 256, 0, stream>>>(y2, n3_dw, n3_db, y3, 128, 7, 128, 4, 2, tot); }
    { int tot = B_SZ * 9 * 16;
      conv1x1_kernel<<<(tot + 255) / 256, 256, 0, stream>>>(y3, n3_tw, n3_tb, rpn + 1470, 128, 16, 9, NUM_ANCH, tot); }

    // 3) NMS + raw feature mean (raw NHWC still live)
    nms_kernel<<<B_SZ, NMS_T, 0, stream>>>(rpn, anch, tidx);
    mean_nhwc_kernel<<<(B_SZ * NCH + 255) / 256, 256, 0, stream>>>(raw, rfeat, B_SZ, 196, NCH);

    // 4) crops -> part images -> backbone conv again (reuses raw buffer)
    { int tot = B_SZ * TOPN * 3 * OUTR * OUTR;
      crop_resize_kernel<<<(tot + 255) / 256, 256, 0, stream>>>(x, anch, tidx, parts); }
    patch_gemm_wmma<<<gemm_grid, 256, 0, stream>>>(parts, bb_w, bb_b, raw, 224, 7, 49);
    mean_nhwc_kernel<<<(B_SZ * TOPN * NCH + 255) / 256, 256, 0, stream>>>(raw, pfeat, B_SZ * TOPN, 49, NCH);

    // 5) classifier
    { int threads = B_SZ * NUM_CLASSES * 32;
      classifier_kernel<<<(threads + 255) / 256, 256, 0, stream>>>(pfeat, rfeat, cn_w, cn_b, out); }
}
